// FlaxPositionWiseMoELayer_85306640433504
// MI455X (gfx1250) — matrix operand order repaired and checked by reading, not compiled, not run on hardware
//
#include <hip/hip_runtime.h>
#include <hip/hip_bf16.h>

// ---------------------------------------------------------------------------
// MoE layer for gfx1250 (MI455X): bf16 WMMA expert GEMMs + gating/dispatch.
// Sizes fixed by the reference: G=2 S=2048 M=768 H=3072 E=16 C=256.
// ---------------------------------------------------------------------------

typedef __bf16 bf16;
typedef bf16  v16bf __attribute__((ext_vector_type(16)));
typedef bf16  v8bf  __attribute__((ext_vector_type(8)));
typedef float v8f   __attribute__((ext_vector_type(8)));

union FragBF { v16bf v; v8bf h[2]; };
union PackBF4 { bf16 h[4]; uint2 u; };   // 4 bf16 -> one 8-byte LDS store

#define MOE_G   2
#define MOE_S   2048
#define MOE_M   768
#define MOE_H   3072
#define MOE_E   16
#define MOE_C   256
#define MOE_GC  (MOE_G * MOE_C)          // 512 rows per expert
#define MOE_T   (MOE_G * MOE_S)          // 4096 tokens
#define MOE_ROWS (MOE_E * MOE_GC)        // 8192 dispatch rows

// ---------------------------------------------------------------------------
// 1) Gating: one wave (32 lanes) per token. logits = x . wg, softmax, top-2.
// ---------------------------------------------------------------------------
__global__ __launch_bounds__(256)
void moe_gating(const float* __restrict__ x, const float* __restrict__ wg,
                int* __restrict__ idx1, int* __restrict__ idx2,
                float* __restrict__ g1, float* __restrict__ g2)
{
    const int lane = threadIdx.x & 31;
    const int wid  = threadIdx.x >> 5;
    const int tok  = blockIdx.x * 8 + wid;          // 4096 tokens total
    const float* xr = x + (long)tok * MOE_M;

    float acc[MOE_E];
    #pragma unroll
    for (int e = 0; e < MOE_E; ++e) acc[e] = 0.f;

    for (int m = lane; m < MOE_M; m += 32) {
        const float xv = xr[m];
        const float4* w4 = (const float4*)(wg + (long)m * MOE_E);
        const float4 a = w4[0], b = w4[1], c = w4[2], d = w4[3];
        acc[0]  += xv * a.x;  acc[1]  += xv * a.y;  acc[2]  += xv * a.z;  acc[3]  += xv * a.w;
        acc[4]  += xv * b.x;  acc[5]  += xv * b.y;  acc[6]  += xv * b.z;  acc[7]  += xv * b.w;
        acc[8]  += xv * c.x;  acc[9]  += xv * c.y;  acc[10] += xv * c.z;  acc[11] += xv * c.w;
        acc[12] += xv * d.x;  acc[13] += xv * d.y;  acc[14] += xv * d.z;  acc[15] += xv * d.w;
    }

    #pragma unroll
    for (int e = 0; e < MOE_E; ++e) {
        #pragma unroll
        for (int off = 16; off > 0; off >>= 1)
            acc[e] += __shfl_xor(acc[e], off, 32);
    }

    if (lane == 0) {
        float mx = acc[0];
        #pragma unroll
        for (int e = 1; e < MOE_E; ++e) mx = fmaxf(mx, acc[e]);
        float p[MOE_E];
        float sum = 0.f;
        #pragma unroll
        for (int e = 0; e < MOE_E; ++e) { p[e] = __expf(acc[e] - mx); sum += p[e]; }
        const float inv = 1.f / sum;
        #pragma unroll
        for (int e = 0; e < MOE_E; ++e) p[e] *= inv;

        int i1 = 0;
        #pragma unroll
        for (int e = 1; e < MOE_E; ++e) if (p[e] > p[i1]) i1 = e;
        const float pv1 = p[i1];
        p[i1] = -1.f;                      // remove top-1 (reference zeroes it)
        int i2 = 0;
        #pragma unroll
        for (int e = 1; e < MOE_E; ++e) if (p[e] > p[i2]) i2 = e;

        idx1[tok] = i1;  idx2[tok] = i2;
        g1[tok] = pv1;   g2[tok] = p[i2];
    }
}

// ---------------------------------------------------------------------------
// 2) Ordered capacity scan (reference cumsum semantics).
// ---------------------------------------------------------------------------
__global__ __launch_bounds__(32)
void moe_scan(const int* __restrict__ idx1, const int* __restrict__ idx2,
              int* __restrict__ r1, int* __restrict__ r2)
{
    const int g = blockIdx.x;
    const int e = threadIdx.x;
    __shared__ int kept[MOE_E];

    if (e < MOE_E) {
        int cnt = 0;
        for (int s = 0; s < MOE_S; ++s) {
            const int t = g * MOE_S + s;
            if (idx1[t] == e) {
                r1[t] = (cnt < MOE_C) ? (e * MOE_GC + g * MOE_C + cnt) : -1;
                ++cnt;
            }
        }
        kept[e] = cnt < MOE_C ? cnt : MOE_C;
    }
    __syncthreads();
    if (e < MOE_E) {
        int cnt = kept[e];
        for (int s = 0; s < MOE_S; ++s) {
            const int t = g * MOE_S + s;
            if (idx2[t] == e) {
                r2[t] = (cnt < MOE_C) ? (e * MOE_GC + g * MOE_C + cnt) : -1;
                ++cnt;
            }
        }
    }
}

// ---------------------------------------------------------------------------
// 3) Normalized combine weights per token.
// ---------------------------------------------------------------------------
__global__ __launch_bounds__(256)
void moe_weights(const float* __restrict__ g1, const float* __restrict__ g2,
                 const int* __restrict__ r1, const int* __restrict__ r2,
                 float* __restrict__ w1, float* __restrict__ w2)
{
    const int t = blockIdx.x * 256 + threadIdx.x;
    const float a = (r1[t] >= 0) ? g1[t] : 0.f;
    const float b = (r2[t] >= 0) ? g2[t] : 0.f;
    float d = a + b;
    d = (d > 0.f) ? d : 1.f;
    w1[t] = a / d;
    w2[t] = b / d;
}

// ---------------------------------------------------------------------------
// 4) Slot -> token map and bf16 dispatch gather.
// ---------------------------------------------------------------------------
__global__ __launch_bounds__(256)
void moe_init_rows(int* __restrict__ tfr)
{
    const int i = blockIdx.x * 256 + threadIdx.x;
    if (i < MOE_ROWS) tfr[i] = -1;
}

__global__ __launch_bounds__(256)
void moe_fill_rows(const int* __restrict__ r1, const int* __restrict__ r2,
                   int* __restrict__ tfr)
{
    const int t = blockIdx.x * 256 + threadIdx.x;
    const int a = r1[t]; if (a >= 0) tfr[a] = t;
    const int b = r2[t]; if (b >= 0) tfr[b] = t;
}

__global__ __launch_bounds__(256)
void moe_dispatch(const float* __restrict__ x, const int* __restrict__ tfr,
                  bf16* __restrict__ disp)
{
    const int row = blockIdx.x;                 // 8192 rows
    const int tok = tfr[row];
    bf16* dr = disp + (long)row * MOE_M;
    if (tok >= 0) {
        const float* xr = x + (long)tok * MOE_M;
        for (int m = threadIdx.x; m < MOE_M; m += 256) dr[m] = (bf16)xr[m];
    } else {
        for (int m = threadIdx.x; m < MOE_M; m += 256) dr[m] = (bf16)0.f;
    }
}

// ---------------------------------------------------------------------------
// 5) bf16 WMMA GEMM with packed LDS staging (see analysis above).
// ---------------------------------------------------------------------------
template <bool RELU, bool OUT_BF16>
__global__ __launch_bounds__(256)
void moe_wmma_gemm(const bf16* __restrict__ Aall, const float* __restrict__ Ball,
                   void* __restrict__ Call,
                   int K, int lda, int ldb, int ldc,
                   long aStride, long bStride, long cStride)
{
    const int e  = blockIdx.z;
    const int mB = blockIdx.y;    // 64-row block
    const int nB = blockIdx.x;    // 256-col block

    const bf16*  A = Aall + (long)e * aStride + (long)mB * 64 * lda;
    const float* B = Ball + (long)e * bStride + (long)nB * 256;

    const int tid  = threadIdx.x;
    const int lane = tid & 31;
    const int wid  = tid >> 5;
    const int wm   = wid & 1;
    const int wn   = wid >> 1;

    __shared__ __attribute__((aligned(16))) bf16 sA[64][40];
    __shared__ __attribute__((aligned(16))) bf16 sB[256][40];

    v8f zacc;
    #pragma unroll
    for (int q = 0; q < 8; ++q) zacc[q] = 0.f;
    v8f acc[2][4];
    #pragma unroll
    for (int i = 0; i < 2; ++i)
        #pragma unroll
        for (int j = 0; j < 4; ++j) acc[i][j] = zacc;

    const int aRow = tid >> 2;            // 0..63
    const int aK   = (tid & 3) * 8;       // 0,8,16,24
    const int bN   = (tid & 63) * 4;      // 0..252 step 4
    const int bK   = (tid >> 6) * 8;      // 0,8,16,24

    const int hi    = lane >> 4;
    const int ll    = lane & 15;
    const int aKoff = hi * 8;
    const int bKoff = hi * 8;

    for (int k0 = 0; k0 < K; k0 += 32) {
        *(v8bf*)&sA[aRow][aK] = *(const v8bf*)(A + (long)aRow * lda + k0 + aK);

        #pragma unroll
        for (int q = 0; q < 2; ++q) {
            const int kk = bK + q * 4;
            const float4 f0 = *(const float4*)(B + (long)(k0 + kk + 0) * ldb + bN);
            const float4 f1 = *(const float4*)(B + (long)(k0 + kk + 1) * ldb + bN);
            const float4 f2 = *(const float4*)(B + (long)(k0 + kk + 2) * ldb + bN);
            const float4 f3 = *(const float4*)(B + (long)(k0 + kk + 3) * ldb + bN);
            PackBF4 p0, p1, p2, p3;
            p0.h[0] = (bf16)f0.x; p0.h[1] = (bf16)f1.x; p0.h[2] = (bf16)f2.x; p0.h[3] = (bf16)f3.x;
            p1.h[0] = (bf16)f0.y; p1.h[1] = (bf16)f1.y; p1.h[2] = (bf16)f2.y; p1.h[3] = (bf16)f3.y;
            p2.h[0] = (bf16)f0.z; p2.h[1] = (bf16)f1.z; p2.h[2] = (bf16)f2.z; p2.h[3] = (bf16)f3.z;
            p3.h[0] = (bf16)f0.w; p3.h[1] = (bf16)f1.w; p3.h[2] = (bf16)f2.w; p3.h[3] = (bf16)f3.w;
            *(uint2*)&sB[bN + 0][kk] = p0.u;
            *(uint2*)&sB[bN + 1][kk] = p1.u;
            *(uint2*)&sB[bN + 2][kk] = p2.u;
            *(uint2*)&sB[bN + 3][kk] = p3.u;
        }
        __syncthreads();

        FragBF a[2], b[4];
        #pragma unroll
        for (int i = 0; i < 2; ++i) {
            const int r = wm * 32 + i * 16 + ll;
            a[i].h[0] = *(const v8bf*)&sA[r][aKoff];
            a[i].h[1] = *(const v8bf*)&sA[r][16 + aKoff];
        }
        #pragma unroll
        for (int j = 0; j < 4; ++j) {
            const int n = wn * 64 + j * 16 + ll;
            b[j].h[0] = *(const v8bf*)&sB[n][bKoff];
            b[j].h[1] = *(const v8bf*)&sB[n][16 + bKoff];
        }

        #pragma unroll
        for (int i = 0; i < 2; ++i)
            #pragma unroll
            for (int j = 0; j < 4; ++j)
                acc[i][j] = __builtin_amdgcn_wmma_f32_16x16x32_bf16(
                    false, a[i].v, false, b[j].v, (short)0, acc[i][j],
                    false, false);
        __syncthreads();
    }

    bf16*  Cb = (bf16*)Call  + (long)e * cStride;
    float* Cf = (float*)Call + (long)e * cStride;
    #pragma unroll
    for (int i = 0; i < 2; ++i) {
        #pragma unroll
        for (int j = 0; j < 4; ++j) {
            const int col = nB * 256 + wn * 64 + j * 16 + ll;
            #pragma unroll
            for (int v = 0; v < 8; ++v) {
                const int row = mB * 64 + wm * 32 + i * 16 + v + 8 * hi;
                float val = acc[i][j][v];
                if (RELU) val = fmaxf(val, 0.f);
                const long off = (long)row * ldc + col;
                if (OUT_BF16) Cb[off] = (bf16)val;
                else          Cf[off] = val;
            }
        }
    }
}

// ---------------------------------------------------------------------------
// 6) Combine: out[t] = w1*eo[r1] + w2*eo[r2]
// ---------------------------------------------------------------------------
__global__ __launch_bounds__(256)
void moe_combine(const float* __restrict__ eo,
                 const int* __restrict__ r1, const int* __restrict__ r2,
                 const float* __restrict__ w1, const float* __restrict__ w2,
                 float* __restrict__ out)
{
    const int t = blockIdx.x;
    const int a = r1[t], b = r2[t];
    const float wa = w1[t], wb = w2[t];
    float* o = out + (long)t * MOE_M;
    for (int m = threadIdx.x; m < MOE_M; m += 256) {
        float v = 0.f;
        if (a >= 0) v += wa * eo[(long)a * MOE_M + m];
        if (b >= 0) v += wb * eo[(long)b * MOE_M + m];
        o[m] = v;
    }
}

// ---------------------------------------------------------------------------
// Launch
// ---------------------------------------------------------------------------
extern "C" void kernel_launch(void* const* d_in, const int* in_sizes, int n_in,
                              void* d_out, int out_size, void* d_ws, size_t ws_size,
                              hipStream_t stream)
{
    const float* x  = (const float*)d_in[0];
    const float* wg = (const float*)d_in[1];
    const float* wi = (const float*)d_in[2];
    const float* wo = (const float*)d_in[3];
    float* out = (float*)d_out;

    char* ws = (char*)d_ws;
    size_t off = 0;
    auto alloc = [&](size_t bytes) { char* p = ws + off; off = (off + bytes + 255) & ~(size_t)255; return p; };

    int*   idx1 = (int*)alloc(MOE_T * 4);
    int*   idx2 = (int*)alloc(MOE_T * 4);
    float* g1   = (float*)alloc(MOE_T * 4);
    float* g2   = (float*)alloc(MOE_T * 4);
    int*   r1   = (int*)alloc(MOE_T * 4);
    int*   r2   = (int*)alloc(MOE_T * 4);
    float* w1   = (float*)alloc(MOE_T * 4);
    float* w2   = (float*)alloc(MOE_T * 4);
    int*   tfr  = (int*)alloc(MOE_ROWS * 4);
    bf16*  disp = (bf16*)alloc((size_t)MOE_ROWS * MOE_M * 2);
    bf16*  h    = (bf16*)alloc((size_t)MOE_E * MOE_GC * MOE_H * 2);
    float* eo   = (float*)alloc((size_t)MOE_E * MOE_GC * MOE_M * 4);
    (void)ws_size;

    moe_gating<<<MOE_T / 8, 256, 0, stream>>>(x, wg, idx1, idx2, g1, g2);
    moe_scan<<<MOE_G, 32, 0, stream>>>(idx1, idx2, r1, r2);
    moe_weights<<<MOE_T / 256, 256, 0, stream>>>(g1, g2, r1, r2, w1, w2);
    moe_init_rows<<<(MOE_ROWS + 255) / 256, 256, 0, stream>>>(tfr);
    moe_fill_rows<<<MOE_T / 256, 256, 0, stream>>>(r1, r2, tfr);
    moe_dispatch<<<MOE_ROWS, 256, 0, stream>>>(x, tfr, disp);

    {
        dim3 grid(MOE_H / 256, MOE_GC / 64, MOE_E);
        moe_wmma_gemm<true, true><<<grid, 256, 0, stream>>>(
            disp, wi, (void*)h,
            MOE_M, MOE_M, MOE_H, MOE_H,
            (long)MOE_GC * MOE_M, (long)MOE_M * MOE_H, (long)MOE_GC * MOE_H);
    }
    {
        dim3 grid(MOE_M / 256, MOE_GC / 64, MOE_E);
        moe_wmma_gemm<false, false><<<grid, 256, 0, stream>>>(
            h, wo, (void*)eo,
            MOE_H, MOE_H, MOE_M, MOE_M,
            (long)MOE_GC * MOE_H, (long)MOE_H * MOE_M, (long)MOE_GC * MOE_M);
    }

    moe_combine<<<MOE_T, 256, 0, stream>>>(eo, r1, r2, w1, w2, out);
}
